// SelfAttentionModel_39848706572956
// MI455X (gfx1250) — matrix operand order repaired and checked by reading, not compiled, not run on hardware
//
#include <hip/hip_runtime.h>
#include <stdint.h>

// Flash-attention for B=2, S=2048, H=16, D=64 on gfx1250 (MI455X).
// FA2 tiling: 32 q-rows per wave (2 subtiles sharing K/V fragments),
// 64-key blocks, cooperative fp32->fp16 K/V staging into LDS shared by 8
// waves; v_wmma_f32_16x16x32_f16 for QK^T, PV and softmax row-sums
// (ones-matrix trick); exp2-domain online softmax with v_permlane16_b32
// butterfly row-max (no LDS round-trips); register-side hash dropout.
// Output fp32 in [B,H,S,D].

#define B_   2
#define S_   2048
#define H_   16
#define D_   64
#define SCALE2_ 0.1803368801111204f   // (1/sqrt(64)) * log2(e)
#define KEEP_T_ 230u                  // keep if byte < 230 (~0.9)
#define KEEP_P_ 0.9f                  // reference dropout keep prob (for 1/(1-p))

typedef __attribute__((ext_vector_type(16))) _Float16 v16h;
typedef __attribute__((ext_vector_type(8)))  _Float16 v8h;
typedef __attribute__((ext_vector_type(8)))  float    v8f;
typedef __attribute__((ext_vector_type(4)))  float    v4f;

// Row-max over each 16-lane group. Preferred path: v_permlane16_b32
// XOR-butterfly (pure VALU, no DS, no wait counters). sel[i] = i ^ m.
__device__ __forceinline__ float rmax16(float v) {
#if defined(__has_builtin) && __has_builtin(__builtin_amdgcn_permlane16)
  unsigned a = __builtin_bit_cast(unsigned, v);
  unsigned t;
  t = __builtin_amdgcn_permlane16(a, a, 0x67452301u, 0xEFCDAB89u, false, false);
  a = __builtin_bit_cast(unsigned,
        fmaxf(__builtin_bit_cast(float, a), __builtin_bit_cast(float, t)));
  t = __builtin_amdgcn_permlane16(a, a, 0x54761032u, 0xDCFE98BAu, false, false);
  a = __builtin_bit_cast(unsigned,
        fmaxf(__builtin_bit_cast(float, a), __builtin_bit_cast(float, t)));
  t = __builtin_amdgcn_permlane16(a, a, 0x32107654u, 0xBA98FEDCu, false, false);
  a = __builtin_bit_cast(unsigned,
        fmaxf(__builtin_bit_cast(float, a), __builtin_bit_cast(float, t)));
  t = __builtin_amdgcn_permlane16(a, a, 0xFEDCBA98u, 0x76543210u, false, false);
  a = __builtin_bit_cast(unsigned,
        fmaxf(__builtin_bit_cast(float, a), __builtin_bit_cast(float, t)));
  return __builtin_bit_cast(float, a);
#else
  v = fmaxf(v, __shfl_xor(v, 1, 32));
  v = fmaxf(v, __shfl_xor(v, 2, 32));
  v = fmaxf(v, __shfl_xor(v, 4, 32));
  v = fmaxf(v, __shfl_xor(v, 8, 32));
  return v;
#endif
}

__device__ __forceinline__ uint32_t hash_u32(uint32_t x) {
  x ^= x >> 16; x *= 0x7feb352du;
  x ^= x >> 15; x *= 0x846ca68bu;
  x ^= x >> 16;
  return x;
}

__device__ __forceinline__ v8h cvt8(v4f x, v4f y) {
  v8h h;
#pragma unroll
  for (int i = 0; i < 4; ++i) { h[i] = (_Float16)x[i]; h[4 + i] = (_Float16)y[i]; }
  return h;
}

// halfs 0..7 from p[0..7], 8..15 from p[16..23] (LDS) -> v16h (two b128 loads)
__device__ __forceinline__ v16h ld_v16h(const _Float16* p) {
  v8h lo = *(const v8h*)(p);
  v8h hi = *(const v8h*)(p + 16);
  return __builtin_shufflevector(lo, hi, 0, 1, 2, 3, 4, 5, 6, 7,
                                 8, 9, 10, 11, 12, 13, 14, 15);
}

// A-fragment: halfs 0..7 from p0[0..7], 8..15 from p1[0..7], scaled fp32 src
__device__ __forceinline__ v16h loadA_cvt(const float* __restrict__ p0,
                                          const float* __restrict__ p1, float s) {
  v4f a = *(const v4f*)(p0);
  v4f b = *(const v4f*)(p0 + 4);
  v4f c = *(const v4f*)(p1);
  v4f d = *(const v4f*)(p1 + 4);
  v16h r;
#pragma unroll
  for (int i = 0; i < 4; ++i) {
    r[i]      = (_Float16)(a[i] * s);
    r[4 + i]  = (_Float16)(b[i] * s);
    r[8 + i]  = (_Float16)(c[i] * s);
    r[12 + i] = (_Float16)(d[i] * s);
  }
  return r;
}

// Zero dropped elements of an A-fragment. One hash -> 4 byte decisions.
__device__ __forceinline__ v16h mask_frag(v16h a, uint32_t ridx, int colbase) {
#pragma unroll
  for (int g = 0; g < 4; ++g) {
    const int cb = colbase + (g >> 1) * 16 + (g & 1) * 4;  // A halfs g*4..g*4+3
    const uint32_t hv = hash_u32(ridx + (uint32_t)cb);
#pragma unroll
    for (int k = 0; k < 4; ++k) {
      if (((hv >> (8 * k)) & 0xffu) >= KEEP_T_) a[g * 4 + k] = (_Float16)0.0f;
    }
  }
  return a;
}

__global__ __launch_bounds__(256) void
fattn_wmma_kernel(const float* __restrict__ Q, const float* __restrict__ K,
                  const float* __restrict__ V, float* __restrict__ O) {
  // Shared: K tile row-major [64 keys][64 d] fp16 (8KB),
  //         V tile transposed [64 d][64 keys] fp16 (8KB),
  //         per-wave P tiles [32 q][64 keys] fp16 (8 x 4KB).
  __shared__ __align__(16) _Float16 ktile[64 * 64];
  __shared__ __align__(16) _Float16 vtile[64 * 64];
  __shared__ __align__(16) _Float16 ptile_all[8 * 32 * 64];

  const int tid  = threadIdx.x;
  const int lane = tid & 31;
  const int wave = tid >> 5;
  const int qblk = blockIdx.x & 7;     // 8 q-blocks of 256 rows per (b,h)
  const int bh   = blockIdx.x >> 3;    // b*H + h
  const int b    = bh >> 4;            // H_ == 16
  const int h    = bh & 15;
  const int q0   = qblk * 256 + wave * 32;   // wave's 32-row q tile

  const int grp = lane >> 4;           // lane group 0/1
  const int ln  = lane & 15;
  const int kb  = grp ? 8 : 0;         // A-layout col offset / C-layout row offset

  _Float16* ptile = ptile_all + wave * (32 * 64);

  // ---- staging thread mappings (cover 64x64 tile with 256 threads) ----
  const int kr = tid >> 2;              // K: row 0..63
  const int kd = (tid & 3) * 16;        //    d segment 0,16,32,48
  const float* kstage = K + (((size_t)b * S_ + kr) * H_ + h) * D_ + kd;
  const int vd = tid & 63;              // V: d column 0..63
  const int vk = (tid >> 6) * 16;       //    key segment 0,16,32,48
  const float* vstage = V + (((size_t)b * S_ + vk) * H_ + h) * D_ + vd;
  const size_t rowpitch = (size_t)H_ * D_;   // floats per key step

  // ---- Q A-fragments, 2 subtiles x 2 d-chunks; exp2-domain scale ----
  v16h aq[2][2];
#pragma unroll
  for (int u = 0; u < 2; ++u) {
    const float* qp = Q + (((size_t)b * S_ + q0 + u * 16 + ln) * H_ + h) * D_;
    aq[u][0] = loadA_cvt(qp + kb,      qp + kb + 16, SCALE2_);      // d [0,32)
    aq[u][1] = loadA_cvt(qp + 32 + kb, qp + kb + 48, SCALE2_);      // d [32,64)
  }

  v16h ones;
#pragma unroll
  for (int i = 0; i < 16; ++i) ones[i] = (_Float16)1.0f;

  float m_run[2][8], l_run[2][8];
  v8f acc[2][4];
#pragma unroll
  for (int u = 0; u < 2; ++u) {
#pragma unroll
    for (int j = 0; j < 8; ++j) { m_run[u][j] = -1e30f; l_run[u][j] = 0.0f; }
#pragma unroll
    for (int t = 0; t < 4; ++t) acc[u][t] = v8f{};
  }

  for (int kb0 = 0; kb0 < S_; kb0 += 64) {
    __syncthreads();   // previous iteration's LDS readers are done

    // ---- cooperative staging: K row-major, V transposed, fp32->fp16 ----
    {
      const float* kp = kstage + (size_t)kb0 * rowpitch;
      v4f k0 = *(const v4f*)(kp);
      v4f k1 = *(const v4f*)(kp + 4);
      v4f k2 = *(const v4f*)(kp + 8);
      v4f k3 = *(const v4f*)(kp + 12);
      *(v8h*)(ktile + kr * 64 + kd)     = cvt8(k0, k1);
      *(v8h*)(ktile + kr * 64 + kd + 8) = cvt8(k2, k3);

      const float* vp = vstage + (size_t)kb0 * rowpitch;
      float vv[16];
#pragma unroll
      for (int i = 0; i < 16; ++i) vv[i] = vp[(size_t)i * rowpitch];
      v4f a = {vv[0],  vv[1],  vv[2],  vv[3]};
      v4f c = {vv[4],  vv[5],  vv[6],  vv[7]};
      v4f e = {vv[8],  vv[9],  vv[10], vv[11]};
      v4f f = {vv[12], vv[13], vv[14], vv[15]};
      *(v8h*)(vtile + vd * 64 + vk)     = cvt8(a, c);
      *(v8h*)(vtile + vd * 64 + vk + 8) = cvt8(e, f);

      if (kb0 + 64 < S_) {   // prefetch next key block (global_prefetch_b8)
        __builtin_prefetch(kp + 64 * rowpitch, 0, 1);
        __builtin_prefetch(vp + 64 * rowpitch, 0, 1);
      }
    }
    __syncthreads();   // staged tiles visible to all waves

    // ---- S = Q K^T : 4 score tiles x 2 q-subtiles; K frags loaded once ----
    v8f s[2][4];
#pragma unroll
    for (int t = 0; t < 4; ++t) {
      const _Float16* kp = ktile + (t * 16 + ln) * 64 + grp * 8;
      v16h bk0 = ld_v16h(kp);        // d chunk [0,32)
      v16h bk1 = ld_v16h(kp + 32);   // d chunk [32,64)
#pragma unroll
      for (int u = 0; u < 2; ++u) {
        v8f z = v8f{};
        z = __builtin_amdgcn_wmma_f32_16x16x32_f16(false, aq[u][0], false, bk0,
                                                   (short)0, z, false, false);
        z = __builtin_amdgcn_wmma_f32_16x16x32_f16(false, aq[u][1], false, bk1,
                                                   (short)0, z, false, false);
        s[u][t] = z;
      }
    }

    // ---- online softmax stats: permlane row max, exp2, P -> LDS ----
    float alpha[2][8];
#pragma unroll
    for (int u = 0; u < 2; ++u) {
#pragma unroll
      for (int j = 0; j < 8; ++j) {
        float tm = fmaxf(fmaxf(s[u][0][j], s[u][1][j]),
                         fmaxf(s[u][2][j], s[u][3][j]));
        tm = rmax16(tm);
        const float m_new = fmaxf(m_run[u][j], tm);
        alpha[u][j] = __builtin_amdgcn_exp2f(m_run[u][j] - m_new);
        m_run[u][j] = m_new;
        const int row = u * 16 + j + kb;
#pragma unroll
        for (int t = 0; t < 4; ++t) {
          const float e = __builtin_amdgcn_exp2f(s[u][t][j] - m_new);
          ptile[row * 64 + t * 16 + ln] = (_Float16)e;   // undropped P
        }
      }
#pragma unroll
      for (int t = 0; t < 4; ++t)
#pragma unroll
        for (int j = 0; j < 8; ++j) acc[u][t][j] *= alpha[u][j];
    }

    // ---- P as A-fragments + WMMA row-sums + register dropout ----
    v16h ap[2][2];
#pragma unroll
    for (int u = 0; u < 2; ++u) {
      const _Float16* pp = ptile + (u * 16 + ln) * 64 + kb;
      v8h a0lo = *(const v8h*)(pp);
      v8h a0hi = *(const v8h*)(pp + 16);
      v8h a1lo = *(const v8h*)(pp + 32);
      v8h a1hi = *(const v8h*)(pp + 48);
      ap[u][0] = __builtin_shufflevector(a0lo, a0hi, 0, 1, 2, 3, 4, 5, 6, 7,
                                         8, 9, 10, 11, 12, 13, 14, 15);
      ap[u][1] = __builtin_shufflevector(a1lo, a1hi, 0, 1, 2, 3, 4, 5, 6, 7,
                                         8, 9, 10, 11, 12, 13, 14, 15);

      v8f rs = v8f{};
      rs = __builtin_amdgcn_wmma_f32_16x16x32_f16(false, ap[u][0], false, ones,
                                                  (short)0, rs, false, false);
      rs = __builtin_amdgcn_wmma_f32_16x16x32_f16(false, ap[u][1], false, ones,
                                                  (short)0, rs, false, false);
#pragma unroll
      for (int j = 0; j < 8; ++j)
        l_run[u][j] = l_run[u][j] * alpha[u][j] + rs[j];

      const uint32_t ridx =
          ((uint32_t)bh * S_ + (uint32_t)(q0 + u * 16 + ln)) * (uint32_t)S_ +
          (uint32_t)kb0;
      ap[u][0] = mask_frag(ap[u][0], ridx, kb);        // cols [0,32)
      ap[u][1] = mask_frag(ap[u][1], ridx, 32 + kb);   // cols [32,64)
    }

    // ---- O += P V : V frags loaded once, shared by both q-subtiles ----
#pragma unroll
    for (int t = 0; t < 4; ++t) {
      const _Float16* vp = vtile + (t * 16 + ln) * 64 + grp * 8;
      v16h bv0 = ld_v16h(vp);        // keys [0,32)
      v16h bv1 = ld_v16h(vp + 32);   // keys [32,64)
#pragma unroll
      for (int u = 0; u < 2; ++u) {
        acc[u][t] = __builtin_amdgcn_wmma_f32_16x16x32_f16(
            false, ap[u][0], false, bv0, (short)0, acc[u][t], false, false);
        acc[u][t] = __builtin_amdgcn_wmma_f32_16x16x32_f16(
            false, ap[u][1], false, bv1, (short)0, acc[u][t], false, false);
      }
    }
  }

  // ---- epilogue: normalize by l, fold 1/(1-p), store [B,H,S,D] fp32 ----
#pragma unroll
  for (int u = 0; u < 2; ++u) {
#pragma unroll
    for (int j = 0; j < 8; ++j) {
      const int row = q0 + u * 16 + j + kb;
      const float sc = 1.0f / (l_run[u][j] * KEEP_P_);
      float* op = O + ((size_t)bh * S_ + row) * D_ + ln;
#pragma unroll
      for (int t = 0; t < 4; ++t)
        op[t * 16] = acc[u][t][j] * sc;
    }
  }
}

extern "C" void kernel_launch(void* const* d_in, const int* in_sizes, int n_in,
                              void* d_out, int out_size, void* d_ws, size_t ws_size,
                              hipStream_t stream) {
  (void)in_sizes; (void)n_in; (void)out_size; (void)d_ws; (void)ws_size;
  const float* Q = (const float*)d_in[0];
  const float* K = (const float*)d_in[1];
  const float* V = (const float*)d_in[2];
  float* O = (float*)d_out;
  // 256 blocks: 32 (b,h) x 8 q-blocks of 256 rows; 256 threads = 8 wave32
  dim3 grid(B_ * H_ * (S_ / 256));
  fattn_wmma_kernel<<<grid, dim3(256), 0, stream>>>(Q, K, V, O);
}
